// VectorQuantizerEMAMLP_20693152432835
// MI455X (gfx1250) — matrix operand order repaired and checked by reading, not compiled, not run on hardware
//
#include <hip/hip_runtime.h>
#include <math.h>

// ---------------------------------------------------------------------------
// VQ-VAE quantizer for MI455X (gfx1250, wave32, WMMA).
// Sizes fixed per reference: B=32, D=256, L=1024, K=4096, N=B*L=32768.
// ---------------------------------------------------------------------------

typedef __attribute__((ext_vector_type(16))) __bf16 v16bf;
typedef __attribute__((ext_vector_type(8)))  float  v8f;

#define BB   32
#define DD   256
#define LL   1024
#define KK   4096
#define NN   32768            // B*L
#define NT   2048             // N/16  (M tiles)
#define KT   256              // K/16  (code tiles)
#define DSN  8                // D/32  (wmma K steps)

// ---- workspace byte offsets --------------------------------------------------
#define WS_A_HI  ((size_t)0)                    // N*D bf16 = 16 MB
#define WS_A_LO  ((size_t)16777216)             // 16 MB
#define WS_B_HI  ((size_t)33554432)             // K*D bf16 = 2 MB
#define WS_B_LO  ((size_t)35651584)             // 2 MB
#define WS_HN    ((size_t)37748736)             // K floats
#define WS_IDX   ((size_t)37765120)             // N ints
#define WS_CNT   ((size_t)37896192)             // K ints
#define WS_PART  ((size_t)37912576)             // N floats (loss partials)
// total ~38.04 MB

// ---- output element offsets (flat float concat, reference return order) -----
#define OUT_LOSS ((size_t)0)
#define OUT_Q    ((size_t)1)                    // [B, D, L] = 8388608
#define OUT_PERP ((size_t)8388609)
#define OUT_EMB  ((size_t)8388610)              // [K, D] = 1048576
#define OUT_IDX  ((size_t)9437186)              // [N, 1]
#define OUT_ENC  ((size_t)9469954)              // [N, K] = 134217728

// ---------------------------------------------------------------------------
// A-matrix 16x32 bf16 lane layout (ISA 7.12.2):
//  lanes 0-15 : M=lane,   VGPR0..3 -> K=0..7,  VGPR4..7 -> K=16..23
//  lanes16-31 : M=lane-16,VGPR0..3 -> K=8..15, VGPR4..7 -> K=24..31
__device__ __forceinline__ int klaneA(int v, int p, int hiHalf) {
    return ((v < 4) ? 0 : 16) + (hiHalf ? 8 : 0) + 2 * (v & 3) + p;
}

// Pack inputs[B,D,L] -> A panels: half index ((t*8+s)*32+lane)*16 + h
__global__ void __launch_bounds__(256)
vq_pack_a(const float* __restrict__ x, __bf16* __restrict__ Ahi,
          __bf16* __restrict__ Alo) {
    int id = blockIdx.x * 256 + threadIdx.x;      // (t,s,lane): one block per tile
    int lane = id & 31, s = (id >> 5) & 7, t = id >> 8;
    int r = lane & 15, hiH = lane >> 4;
    int n = t * 16 + r;
    int b = n >> 10, li = n & 1023;
    size_t obase = (size_t)id * 16;
#pragma unroll
    for (int h = 0; h < 16; ++h) {
        int v = h >> 1, p = h & 1;
        int d = s * 32 + klaneA(v, p, hiH);
        float xv = x[((size_t)(b * DD + d)) * LL + li];
        __bf16 hi = (__bf16)xv;
        Ahi[obase + h] = hi;
        Alo[obase + h] = (__bf16)(xv - (float)hi);
    }
}

// B-matrix 32x16 bf16 lane layout: same K order as A (lanes 0-15: K=0..7,16..23;
// lanes 16-31: K=8..15,24..31); column n = lane&15.
__global__ void __launch_bounds__(256)
vq_pack_b(const float* __restrict__ e, __bf16* __restrict__ Bhi,
          __bf16* __restrict__ Blo) {
    int id = blockIdx.x * 256 + threadIdx.x;      // (tc,s,lane): one block per tile
    int lane = id & 31, s = (id >> 5) & 7, tc = id >> 8;
    int c = tc * 16 + (lane & 15);
    int hiH = lane >> 4;
    size_t obase = (size_t)id * 16;
#pragma unroll
    for (int h = 0; h < 16; ++h) {
        int v = h >> 1, p = h & 1;
        int d = s * 32 + klaneA(v, p, hiH);
        float ev = e[(size_t)c * DD + d];
        __bf16 hi = (__bf16)ev;
        Bhi[obase + h] = hi;
        Blo[obase + h] = (__bf16)(ev - (float)hi);
    }
}

__global__ void __launch_bounds__(256)
vq_halfnorm(const float* __restrict__ e, float* __restrict__ hn) {
    int c = blockIdx.x * 256 + threadIdx.x;
    const float4* row = (const float4*)(e + (size_t)c * DD);
    float s = 0.f;
#pragma unroll
    for (int d = 0; d < DD / 4; ++d) {
        float4 v = row[d];
        s += v.x * v.x + v.y * v.y + v.z * v.z + v.w * v.w;
    }
    hn[c] = 0.5f * s;
}

__global__ void __launch_bounds__(256)
vq_zero(int* __restrict__ counts) {
    int i = blockIdx.x * 256 + threadIdx.x;
    if (i < KK) counts[i] = 0;
}

// ---------------------------------------------------------------------------
// Main kernel: distances via bf16 hi/lo WMMA + per-row argmin over K codes.
// 128 threads = 4 waves; wave w owns M-tile t = blockIdx*4 + w.
// B tile (16 codes x 256 dims, hi+lo = 16 KB) staged in LDS per iteration.
// Three independent accumulator chains (hi*hi, hi*lo, lo*hi) provide ILP=3
// inside the wave so back-to-back WMMAs don't stall on the XDL pipeline
// (occupancy is ~1 wave/SIMD at this VGPR count, so cross-wave ILP is absent).
// ---------------------------------------------------------------------------
__global__ void __launch_bounds__(128)
vq_gemm_argmin(const __bf16* __restrict__ Ahi, const __bf16* __restrict__ Alo,
               const __bf16* __restrict__ Bhi, const __bf16* __restrict__ Blo,
               const float* __restrict__ hn, int* __restrict__ outIdx,
               int* __restrict__ counts) {
    __shared__ __align__(32) __bf16 sB[2 * DSN * 32 * 16];   // 16 KB
    const int tid  = threadIdx.x;
    const int lane = tid & 31;
    const int wv   = tid >> 5;
    const int t    = blockIdx.x * 4 + wv;

    // Resident A panel: 8 D-steps x (hi,lo) = 128 VGPRs.
    v16bf ahr[DSN], alr[DSN];
#pragma unroll
    for (int s = 0; s < DSN; ++s) {
        size_t off = ((size_t)(t * DSN + s) * 32 + lane) * 16;
        ahr[s] = *(const v16bf*)(Ahi + off);
        alr[s] = *(const v16bf*)(Alo + off);
    }

    float best[8];
    int   bidx[8];
#pragma unroll
    for (int j = 0; j < 8; ++j) { best[j] = 3.4e38f; bidx[j] = 0; }

    const uint4* gBh = (const uint4*)Bhi;      // 512 uint4 per code tile
    const uint4* gBl = (const uint4*)Blo;
    uint4* sB4 = (uint4*)sB;

    for (int tc = 0; tc < KT; ++tc) {
        // stage 1024 uint4 (hi then lo) with 128 threads x 8
#pragma unroll
        for (int i = 0; i < 4; ++i) {
            int u = tid + 128 * i;
            sB4[u]       = gBh[tc * 512 + u];
            sB4[512 + u] = gBl[tc * 512 + u];
        }
        if (tc + 1 < KT) {   // pull next tile toward L2/L0 (global_prefetch_b8)
            __builtin_prefetch(gBh + (tc + 1) * 512 + (tid << 2), 0, 1);
            __builtin_prefetch(gBl + (tc + 1) * 512 + (tid << 2), 0, 1);
        }
        __syncthreads();

        v8f acc_hh = {};   // hi(x) * hi(e)
        v8f acc_hl = {};   // hi(x) * lo(e)
        v8f acc_lh = {};   // lo(x) * hi(e)
#pragma unroll
        for (int s = 0; s < DSN; ++s) {
            v16bf bh = *(const v16bf*)(sB + (s * 32 + lane) * 16);
            v16bf bl = *(const v16bf*)(sB + 4096 + (s * 32 + lane) * 16);
            acc_hh = __builtin_amdgcn_wmma_f32_16x16x32_bf16(
                false, ahr[s], false, bh, (short)0, acc_hh, false, false);
            acc_hl = __builtin_amdgcn_wmma_f32_16x16x32_bf16(
                false, ahr[s], false, bl, (short)0, acc_hl, false, false);
            acc_lh = __builtin_amdgcn_wmma_f32_16x16x32_bf16(
                false, alr[s], false, bh, (short)0, acc_lh, false, false);
        }

        // score = 0.5*||e||^2 - x.e (same argmin as full squared distance)
        const int   c = tc * 16 + (lane & 15);
        const float h = hn[c];
#pragma unroll
        for (int j = 0; j < 8; ++j) {
            float sc = h - (acc_hh[j] + (acc_hl[j] + acc_lh[j]));
            if (sc < best[j]) { best[j] = sc; bidx[j] = c; }
        }
        __syncthreads();
    }

    // reduce across the 16 lanes holding the same row set (offsets stay in half)
#pragma unroll
    for (int j = 0; j < 8; ++j) {
#pragma unroll
        for (int off = 1; off < 16; off <<= 1) {
            float ob = __shfl_xor(best[j], off, 32);
            int   oi = __shfl_xor(bidx[j], off, 32);
            if (ob < best[j] || (ob == best[j] && oi < bidx[j])) {
                best[j] = ob; bidx[j] = oi;
            }
        }
    }
    if ((lane & 15) == 0) {
        int rowbase = t * 16 + ((lane >> 4) ? 8 : 0);   // lane0: M0..7, lane16: M8..15
#pragma unroll
        for (int j = 0; j < 8; ++j) {
            outIdx[rowbase + j] = bidx[j];
            atomicAdd(&counts[bidx[j]], 1);
        }
    }
}

// ---------------------------------------------------------------------------
// Outputs
// ---------------------------------------------------------------------------
__global__ void __launch_bounds__(256)
vq_quant_loss(const float* __restrict__ x, const float* __restrict__ e,
              const int* __restrict__ idx, float* __restrict__ out,
              float* __restrict__ part) {
    __shared__ float red[256];
    int n = blockIdx.x, d = threadIdx.x;
    int k = idx[n];
    int b = n >> 10, li = n & 1023;
    float q  = e[(size_t)k * DD + d];
    size_t xoff = ((size_t)(b * DD + d)) * LL + li;   // [B,D,L] layout
    float xv = x[xoff];
    out[OUT_Q + xoff] = q;                            // straight-through value == q
    float df = q - xv;
    red[d] = df * df;
    __syncthreads();
    for (int st = 128; st > 0; st >>= 1) {
        if (d < st) red[d] += red[d + st];
        __syncthreads();
    }
    if (d == 0) part[n] = red[0];
}

__global__ void __launch_bounds__(256)
vq_copy_emb(const float* __restrict__ e, float* __restrict__ out) {
    size_t i = (size_t)blockIdx.x * 256 + threadIdx.x;
    out[OUT_EMB + i] = e[i];
}

__global__ void __launch_bounds__(256)
vq_idx_out(const int* __restrict__ idx, float* __restrict__ out) {
    int i = blockIdx.x * 256 + threadIdx.x;
    out[OUT_IDX + i] = (float)idx[i];
}

__global__ void __launch_bounds__(256)
vq_onehot(const int* __restrict__ idx, float* __restrict__ out) {
    int n = blockIdx.x;
    int k = idx[n];
    size_t base = OUT_ENC + (size_t)n * KK;
#pragma unroll
    for (int i = 0; i < 16; ++i) {
        int kk = i * 256 + threadIdx.x;
        __builtin_nontemporal_store(kk == k ? 1.0f : 0.0f, &out[base + kk]);
    }
}

__global__ void __launch_bounds__(256)
vq_scalars(const float* __restrict__ part, const int* __restrict__ counts,
           float* __restrict__ out) {
    __shared__ float red[256];
    int t = threadIdx.x;
    float s = 0.f;
    for (int i = 0; i < 128; ++i) s += part[t * 128 + i];   // fixed order
    red[t] = s;
    __syncthreads();
    for (int st = 128; st > 0; st >>= 1) {
        if (t < st) red[t] += red[t + st];
        __syncthreads();
    }
    float sse = red[0];
    __syncthreads();
    float hsum = 0.f;
    for (int i = 0; i < 16; ++i) {
        float p = (float)counts[t * 16 + i] * (1.0f / (float)NN);
        hsum += p * logf(p + 1e-10f);
    }
    red[t] = hsum;
    __syncthreads();
    for (int st = 128; st > 0; st >>= 1) {
        if (t < st) red[t] += red[t + st];
        __syncthreads();
    }
    if (t == 0) {
        out[OUT_LOSS] = 0.25f * sse / ((float)NN * (float)DD);
        out[OUT_PERP] = expf(-red[0]);
    }
}

// ---------------------------------------------------------------------------
extern "C" void kernel_launch(void* const* d_in, const int* in_sizes, int n_in,
                              void* d_out, int out_size, void* d_ws, size_t ws_size,
                              hipStream_t stream) {
    const float* x   = (const float*)d_in[0];   // inputs [B, D, L]
    const float* emb = (const float*)d_in[1];   // embedding [K, D]
    float* out = (float*)d_out;
    char* ws = (char*)d_ws;

    __bf16* Ahi = (__bf16*)(ws + WS_A_HI);
    __bf16* Alo = (__bf16*)(ws + WS_A_LO);
    __bf16* Bhi = (__bf16*)(ws + WS_B_HI);
    __bf16* Blo = (__bf16*)(ws + WS_B_LO);
    float*  hn  = (float*)(ws + WS_HN);
    int*    idx = (int*)(ws + WS_IDX);
    int*    cnt = (int*)(ws + WS_CNT);
    float*  prt = (float*)(ws + WS_PART);

    vq_zero<<<KK / 256, 256, 0, stream>>>(cnt);
    vq_pack_a<<<NT, 256, 0, stream>>>(x, Ahi, Alo);
    vq_pack_b<<<KT, 256, 0, stream>>>(emb, Bhi, Blo);
    vq_halfnorm<<<KK / 256, 256, 0, stream>>>(emb, hn);

    vq_gemm_argmin<<<NT / 4, 128, 0, stream>>>(Ahi, Alo, Bhi, Blo, hn, idx, cnt);

    vq_quant_loss<<<NN, 256, 0, stream>>>(x, emb, idx, out, prt);
    vq_copy_emb<<<(KK * DD) / 256, 256, 0, stream>>>(emb, out);
    vq_idx_out<<<NN / 256, 256, 0, stream>>>(idx, out);
    vq_onehot<<<NN, 256, 0, stream>>>(idx, out);
    vq_scalars<<<1, 256, 0, stream>>>(prt, cnt, out);
}
